// AudioQuantizer_45320494907628
// MI455X (gfx1250) — matrix operand order repaired and checked by reading, not compiled, not run on hardware
//
#include <hip/hip_runtime.h>

// ---------------------------------------------------------------------------
// Vector quantization: for each of N=16384 rows of x[.,512], find argmin_k
// ||x - codebook[k]||^2 over K=8192 codes, output codebook[argmin] (fp32).
//
// argmin ||x-c||^2  ==  argmax (x.c - 0.5*||c||^2)   (||x||^2 is constant)
//
// Compute-bound (137 GFLOP vs ~80MB HBM) -> bf16 WMMA (v_wmma_f32_16x16x32_bf16).
// Phase 1 converts the codebook to bf16 once (d_ws) + exact fp32 ||c||^2.
// Phase 2 hot loop: async global->LDS DMA of the next 32-row codebook chunk
// (ASYNCcnt, immediate offsets on one base address pair) overlapped with 16
// WMMAs on the current chunk, double-buffered, one barrier per chunk.
// ---------------------------------------------------------------------------

typedef __attribute__((ext_vector_type(16))) __bf16        v16bf;
typedef __attribute__((ext_vector_type(8)))  float         v8f;
typedef __attribute__((ext_vector_type(4)))  unsigned int  v4u;
typedef int v4i __attribute__((vector_size(16)));   // matches async-LDS builtin param

#define GLOBAL_AS __attribute__((address_space(1)))
#define LDS_AS    __attribute__((address_space(3)))

#if defined(__has_builtin)
#  if __has_builtin(__builtin_amdgcn_global_load_async_to_lds_b128)
#    define USE_ASYNC_LDS 1
#  endif
#endif
#ifndef USE_ASYNC_LDS
#  define USE_ASYNC_LDS 0
#endif

#if USE_ASYNC_LDS
// 8 x b128 = 128B burst; IOFFSET applies to BOTH global and LDS addresses
// so one addr pair feeds all 8 loads.
#define STAGE_128B(gp, lp)                                                \
  do {                                                                    \
    __builtin_amdgcn_global_load_async_to_lds_b128((gp), (lp),   0, 0);   \
    __builtin_amdgcn_global_load_async_to_lds_b128((gp), (lp),  16, 0);   \
    __builtin_amdgcn_global_load_async_to_lds_b128((gp), (lp),  32, 0);   \
    __builtin_amdgcn_global_load_async_to_lds_b128((gp), (lp),  48, 0);   \
    __builtin_amdgcn_global_load_async_to_lds_b128((gp), (lp),  64, 0);   \
    __builtin_amdgcn_global_load_async_to_lds_b128((gp), (lp),  80, 0);   \
    __builtin_amdgcn_global_load_async_to_lds_b128((gp), (lp),  96, 0);   \
    __builtin_amdgcn_global_load_async_to_lds_b128((gp), (lp), 112, 0);   \
  } while (0)
#endif

__device__ __forceinline__ void wait_async_zero() {
#if defined(__has_builtin)
#  if __has_builtin(__builtin_amdgcn_s_wait_asynccnt)
  __builtin_amdgcn_s_wait_asynccnt(0);
  return;
#  else
  asm volatile("s_wait_asynccnt 0" ::: "memory");
  return;
#  endif
#else
  asm volatile("s_wait_asynccnt 0" ::: "memory");
#endif
}

union AFrag { unsigned int u[8]; v16bf v; };
union BFrag { v4u q[2];          v16bf v; };

__device__ __forceinline__ unsigned int pack_bf16x2(float a, float b) {
  // round-to-nearest-even f32 -> bf16, packed low|high
  unsigned int ua = __float_as_uint(a);
  unsigned int ub = __float_as_uint(b);
  ua = (ua + 0x7FFFu + ((ua >> 16) & 1u)) >> 16;
  ub = (ub + 0x7FFFu + ((ub >> 16) & 1u)) >> 16;
  return ua | (ub << 16);
}

// ---------------------------------------------------------------------------
// Phase 1: codebook f32 -> bf16 (row-major, unpadded) + fp32 ||c||^2 per row.
// One wave per codebook row; lane covers 16 consecutive dims.
// ---------------------------------------------------------------------------
__global__ __launch_bounds__(256)
void vq_prep(const float* __restrict__ cbk,
             unsigned short* __restrict__ cb16,
             float* __restrict__ csq) {
  const int lane = threadIdx.x & 31;
  const int row  = blockIdx.x * 8 + (threadIdx.x >> 5);
  const float* src = cbk + (size_t)row * 512 + lane * 16;
  float4 q0 = ((const float4*)src)[0];
  float4 q1 = ((const float4*)src)[1];
  float4 q2 = ((const float4*)src)[2];
  float4 q3 = ((const float4*)src)[3];
  float ss = q0.x*q0.x + q0.y*q0.y + q0.z*q0.z + q0.w*q0.w
           + q1.x*q1.x + q1.y*q1.y + q1.z*q1.z + q1.w*q1.w
           + q2.x*q2.x + q2.y*q2.y + q2.z*q2.z + q2.w*q2.w
           + q3.x*q3.x + q3.y*q3.y + q3.z*q3.z + q3.w*q3.w;
  v4u d0 = { pack_bf16x2(q0.x,q0.y), pack_bf16x2(q0.z,q0.w),
             pack_bf16x2(q1.x,q1.y), pack_bf16x2(q1.z,q1.w) };
  v4u d1 = { pack_bf16x2(q2.x,q2.y), pack_bf16x2(q2.z,q2.w),
             pack_bf16x2(q3.x,q3.y), pack_bf16x2(q3.z,q3.w) };
  v4u* dst = (v4u*)(cb16 + (size_t)row * 512 + lane * 16);
  dst[0] = d0;
  dst[1] = d1;
  #pragma unroll
  for (int off = 16; off >= 1; off >>= 1) ss += __shfl_xor(ss, off, 32);
  if (lane == 0) csq[row] = ss;
}

// ---------------------------------------------------------------------------
// Phase 2: WMMA argmax + gather. 64 x-rows per 256-thread block (8 waves:
// 4 M-groups x 2 N-groups), codebook streamed in 32-row chunks, double-buffered.
// ---------------------------------------------------------------------------
__global__ __launch_bounds__(256)
void vq_main(const float* __restrict__ x,
             const float* __restrict__ cbk,           // fp32, for final gather
             const unsigned short* __restrict__ cb16, // bf16 codebook
             const float* __restrict__ csq,           // fp32 ||c||^2
             float* __restrict__ out,
             int K) {
  constexpr int D     = 512;
  constexpr int ROWS  = 64;
  constexpr int CHUNK = 32;
  constexpr int CBS   = 520;  // padded LDS row stride (bf16 elems): 4-bank rotate/row

  __shared__ unsigned short cb[2][CHUNK * CBS];  // 2 x 33,280 B, double buffer

  const int tid  = threadIdx.x;
  const int lane = tid & 31;
  const int wave = tid >> 5;
  const int mgrp = wave & 3;
  const int ngrp = wave >> 2;
  const int hsel = lane >> 4;
  const int l15  = lane & 15;
  const int row0 = blockIdx.x * ROWS;

  // ---- Preload A fragments (16 x-rows x 512 dims, bf16, WMMA A layout) ----
  // 16-bit A 16x32: lanes 0-15 (M=lane): elems 0..7 = K0..7, 8..15 = K16..23
  //                 lanes 16-31        : elems 0..7 = K8..15, 8..15 = K24..31
  const float* xrow = x + (size_t)(row0 + mgrp * 16 + l15) * D;
  AFrag a[16];
  #pragma unroll
  for (int s = 0; s < 16; ++s) {
    const float* p = xrow + s * 32 + hsel * 8;
    float4 q0 = ((const float4*)p)[0];
    float4 q1 = ((const float4*)p)[1];
    float4 q2 = ((const float4*)(p + 16))[0];
    float4 q3 = ((const float4*)(p + 16))[1];
    a[s].u[0] = pack_bf16x2(q0.x, q0.y);
    a[s].u[1] = pack_bf16x2(q0.z, q0.w);
    a[s].u[2] = pack_bf16x2(q1.x, q1.y);
    a[s].u[3] = pack_bf16x2(q1.z, q1.w);
    a[s].u[4] = pack_bf16x2(q2.x, q2.y);
    a[s].u[5] = pack_bf16x2(q2.z, q2.w);
    a[s].u[6] = pack_bf16x2(q3.x, q3.y);
    a[s].u[7] = pack_bf16x2(q3.z, q3.w);
  }

  float bval[8];
  int   bidx[8];
  #pragma unroll
  for (int r = 0; r < 8; ++r) { bval[r] = -3.4e38f; bidx[r] = 0; }

  // staging mapping: 8 threads per chunk row, 128B (8 x b128) per thread
  const int trow = tid >> 3;
  const int tseg = tid & 7;
  const size_t gseg = (size_t)trow * D + tseg * 64;     // elem offset inside chunk
  unsigned short* ldst0 = &cb[0][trow * CBS + tseg * 64];
  unsigned short* ldst1 = &cb[1][trow * CBS + tseg * 64];
  const unsigned short* brow0 = &cb[0][(ngrp * 16 + l15) * CBS];
  const unsigned short* brow1 = &cb[1][(ngrp * 16 + l15) * CBS];

  const int nchunks = K / CHUNK;

  // ---- prologue: stage chunk 0 into buffer 0 ----
#if USE_ASYNC_LDS
  {
    GLOBAL_AS v4i* g = (GLOBAL_AS v4i*)(cb16 + gseg);
    LDS_AS v4i*    l = (LDS_AS v4i*)ldst0;
    STAGE_128B(g, l);
    wait_async_zero();
  }
#else
  {
    const v4u* src = (const v4u*)(cb16 + gseg);
    v4u* dst = (v4u*)ldst0;
    #pragma unroll
    for (int j = 0; j < 8; ++j) dst[j] = src[j];
  }
#endif
  __syncthreads();

  for (int ch = 0; ch < nchunks; ++ch) {
    const int cur = ch & 1;
    const int chn = (ch + 1 < nchunks) ? ch + 1 : ch;

#if USE_ASYNC_LDS
    // ---- fire async DMA of next chunk into the other buffer (overlaps WMMAs) ----
    {
      GLOBAL_AS v4i* g = (GLOBAL_AS v4i*)(cb16 + (size_t)chn * CHUNK * D + gseg);
      LDS_AS v4i*    l = (LDS_AS v4i*)(cur ? ldst0 : ldst1);
      STAGE_128B(g, l);
    }
#else
    v4u nx[8];
    {
      const v4u* src = (const v4u*)(cb16 + (size_t)chn * CHUNK * D + gseg);
      #pragma unroll
      for (int j = 0; j < 8; ++j) nx[j] = src[j];
    }
#endif

    // ---- 16x16x512 dot tile: 16 WMMAs over D ----
    // B 32x16 layout: lane = column (l15); lanes 0-15 hold K0..7, K16..23,
    // lanes 16-31 hold K8..15, K24..31 (matches A table order).
    const unsigned short* brow = cur ? brow1 : brow0;
    v8f acc = {0.f, 0.f, 0.f, 0.f, 0.f, 0.f, 0.f, 0.f};
    #pragma unroll
    for (int s = 0; s < 16; ++s) {
      BFrag b;
      const v4u* bp = (const v4u*)(brow + s * 32 + hsel * 8);
      b.q[0] = bp[0];
      b.q[1] = bp[2];
      acc = __builtin_amdgcn_wmma_f32_16x16x32_bf16(
          false, a[s].v, false, b.v, (short)0, acc, false, false);
    }

    // ---- running argmax update (strict > keeps lowest index per lane) ----
    const int   n        = ch * CHUNK + ngrp * 16 + l15;
    const float half_csq = 0.5f * csq[n];
    #pragma unroll
    for (int r = 0; r < 8; ++r) {
      float sc = acc[r] - half_csq;
      if (sc > bval[r]) { bval[r] = sc; bidx[r] = n; }
    }

#if USE_ASYNC_LDS
    wait_async_zero();                 // our DMA into the other buffer is done
#else
    {
      v4u* dst = (v4u*)(cur ? ldst0 : ldst1);
      #pragma unroll
      for (int j = 0; j < 8; ++j) dst[j] = nx[j];
    }
#endif
    __syncthreads();
  }

  // ---- cross-lane tournament within each 16-lane half ----
  #pragma unroll
  for (int off = 8; off >= 1; off >>= 1) {
    #pragma unroll
    for (int r = 0; r < 8; ++r) {
      float ov = __shfl_xor(bval[r], off, 32);
      int   oi = __shfl_xor(bidx[r], off, 32);
      if (ov > bval[r] || (ov == bval[r] && oi < bidx[r])) { bval[r] = ov; bidx[r] = oi; }
    }
  }

  // reduction scratch overlaid on the (now dead) staging buffer
  float* red_val = (float*)&cb[0][0];          // 128 floats
  int*   red_idx = ((int*)&cb[0][0]) + 128;    // 128 ints
  int*   fidx    = ((int*)&cb[0][0]) + 256;    // 64 ints

  if (l15 == 0) {
    #pragma unroll
    for (int r = 0; r < 8; ++r) {
      const int m = mgrp * 16 + r + 8 * hsel;  // C layout: lane half -> M = r / r+8
      red_val[m * 2 + ngrp] = bval[r];
      red_idx[m * 2 + ngrp] = bidx[r];
    }
  }
  __syncthreads();

  if (tid < ROWS) {
    float v0 = red_val[tid * 2 + 0], v1 = red_val[tid * 2 + 1];
    int   i0 = red_idx[tid * 2 + 0], i1 = red_idx[tid * 2 + 1];
    fidx[tid] = (v1 > v0 || (v1 == v0 && i1 < i0)) ? i1 : i0;
  }
  __syncthreads();

  // ---- gather winning fp32 codebook rows ----
  for (int idx = tid; idx < ROWS * (D / 4); idx += 256) {
    const int row = idx >> 7;
    const int c4  = idx & 127;
    float4 v = ((const float4*)(cbk + (size_t)fidx[row] * D))[c4];
    ((float4*)(out + (size_t)(row0 + row) * D))[c4] = v;
  }
}

// ---------------------------------------------------------------------------
// Fallback (no workspace): fused convert-in-loop version.
// ---------------------------------------------------------------------------
__global__ __launch_bounds__(256)
void vq_fused(const float* __restrict__ x,
              const float* __restrict__ cbk,
              float* __restrict__ out,
              int K) {
  constexpr int D     = 512;
  constexpr int ROWS  = 64;
  constexpr int CHUNK = 32;
  constexpr int CBS   = 520;

  __shared__ unsigned short cb[CHUNK * CBS];
  __shared__ float          csql[CHUNK];
  __shared__ float          red_val[ROWS * 2];
  __shared__ int            red_idx[ROWS * 2];
  __shared__ int            fidx[ROWS];

  const int tid  = threadIdx.x;
  const int lane = tid & 31;
  const int wave = tid >> 5;
  const int mgrp = wave & 3;
  const int ngrp = wave >> 2;
  const int hsel = lane >> 4;
  const int l15  = lane & 15;
  const int row0 = blockIdx.x * ROWS;

  const float* xrow = x + (size_t)(row0 + mgrp * 16 + l15) * D;
  AFrag a[16];
  #pragma unroll
  for (int s = 0; s < 16; ++s) {
    const float* p = xrow + s * 32 + hsel * 8;
    float4 q0 = ((const float4*)p)[0];
    float4 q1 = ((const float4*)p)[1];
    float4 q2 = ((const float4*)(p + 16))[0];
    float4 q3 = ((const float4*)(p + 16))[1];
    a[s].u[0] = pack_bf16x2(q0.x, q0.y);
    a[s].u[1] = pack_bf16x2(q0.z, q0.w);
    a[s].u[2] = pack_bf16x2(q1.x, q1.y);
    a[s].u[3] = pack_bf16x2(q1.z, q1.w);
    a[s].u[4] = pack_bf16x2(q2.x, q2.y);
    a[s].u[5] = pack_bf16x2(q2.z, q2.w);
    a[s].u[6] = pack_bf16x2(q3.x, q3.y);
    a[s].u[7] = pack_bf16x2(q3.z, q3.w);
  }

  float bval[8];
  int   bidx[8];
  #pragma unroll
  for (int r = 0; r < 8; ++r) { bval[r] = -3.4e38f; bidx[r] = 0; }

  const unsigned short* brow = &cb[(ngrp * 16 + l15) * CBS];
  const int nchunks = K / CHUNK;

  for (int ch = 0; ch < nchunks; ++ch) {
    __syncthreads();
    if (tid < CHUNK) csql[tid] = 0.0f;
    __syncthreads();
    {
      const int trow = tid >> 3;
      const int tseg = tid & 7;
      const float* src = cbk + (size_t)(ch * CHUNK + trow) * D + tseg * 64;
      unsigned short* dst = &cb[trow * CBS + tseg * 64];
      float ss = 0.0f;
      #pragma unroll
      for (int j = 0; j < 16; ++j) {
        float4 q = ((const float4*)src)[j];
        ss += q.x * q.x + q.y * q.y + q.z * q.z + q.w * q.w;
        ((uint2*)dst)[j] = make_uint2(pack_bf16x2(q.x, q.y), pack_bf16x2(q.z, q.w));
      }
      atomicAdd(&csql[trow], ss);
    }
    __syncthreads();

    v8f acc = {0.f, 0.f, 0.f, 0.f, 0.f, 0.f, 0.f, 0.f};
    #pragma unroll
    for (int s = 0; s < 16; ++s) {
      BFrag b;
      const v4u* bp = (const v4u*)(brow + s * 32 + hsel * 8);
      b.q[0] = bp[0];
      b.q[1] = bp[2];
      acc = __builtin_amdgcn_wmma_f32_16x16x32_bf16(
          false, a[s].v, false, b.v, (short)0, acc, false, false);
    }

    const int   n        = ch * CHUNK + ngrp * 16 + l15;
    const float half_csq = 0.5f * csql[ngrp * 16 + l15];
    #pragma unroll
    for (int r = 0; r < 8; ++r) {
      float sc = acc[r] - half_csq;
      if (sc > bval[r]) { bval[r] = sc; bidx[r] = n; }
    }
  }

  #pragma unroll
  for (int off = 8; off >= 1; off >>= 1) {
    #pragma unroll
    for (int r = 0; r < 8; ++r) {
      float ov = __shfl_xor(bval[r], off, 32);
      int   oi = __shfl_xor(bidx[r], off, 32);
      if (ov > bval[r] || (ov == bval[r] && oi < bidx[r])) { bval[r] = ov; bidx[r] = oi; }
    }
  }
  if (l15 == 0) {
    #pragma unroll
    for (int r = 0; r < 8; ++r) {
      const int m = mgrp * 16 + r + 8 * hsel;
      red_val[m * 2 + ngrp] = bval[r];
      red_idx[m * 2 + ngrp] = bidx[r];
    }
  }
  __syncthreads();
  if (tid < ROWS) {
    float v0 = red_val[tid * 2 + 0], v1 = red_val[tid * 2 + 1];
    int   i0 = red_idx[tid * 2 + 0], i1 = red_idx[tid * 2 + 1];
    fidx[tid] = (v1 > v0 || (v1 == v0 && i1 < i0)) ? i1 : i0;
  }
  __syncthreads();
  for (int idx = tid; idx < ROWS * (D / 4); idx += 256) {
    const int row = idx >> 7;
    const int c4  = idx & 127;
    float4 v = ((const float4*)(cbk + (size_t)fidx[row] * D))[c4];
    ((float4*)(out + (size_t)(row0 + row) * D))[c4] = v;
  }
}

extern "C" void kernel_launch(void* const* d_in, const int* in_sizes, int n_in,
                              void* d_out, int out_size, void* d_ws, size_t ws_size,
                              hipStream_t stream) {
  const float* x   = (const float*)d_in[0];   // [8,2048,512] f32 -> [16384][512]
  const float* cbk = (const float*)d_in[1];   // [8192,512] f32
  float*       out = (float*)d_out;           // [16384][512] f32

  const int D = 512;
  const int N = in_sizes[0] / D;              // 16384
  const int K = in_sizes[1] / D;              // 8192
  const int blocks = N / 64;

  const size_t need = (size_t)K * D * sizeof(unsigned short)  // bf16 codebook
                    + (size_t)K * sizeof(float);              // ||c||^2

  if (ws_size >= need) {
    unsigned short* cb16 = (unsigned short*)d_ws;
    float*          csq  = (float*)((char*)d_ws + (size_t)K * D * sizeof(unsigned short));
    vq_prep<<<K / 8, 256, 0, stream>>>(cbk, cb16, csq);
    vq_main<<<blocks, 256, 0, stream>>>(x, cbk, cb16, csq, out, K);
  } else {
    vq_fused<<<blocks, 256, 0, stream>>>(x, cbk, out, K);
  }
}
